// PointCloudCorrector_42614665511157
// MI455X (gfx1250) — matrix operand order repaired and checked by reading, not compiled, not run on hardware
//
#include <hip/hip_runtime.h>
#include <hip/hip_bf16.h>
#include <math.h>

typedef __attribute__((ext_vector_type(16))) __bf16 v16bf;
typedef __attribute__((ext_vector_type(8)))  __bf16 v8bf;
typedef __attribute__((ext_vector_type(8)))  float  v8f;

#define C_IN   384
#define C_OUT  64
#define HW     128
#define KSTEP  32
#define PADC   40            // 80B pixel stride in As: 20 banks, conflict-free b128
#define PADK   40            // 80B row stride in Bs
#define AS_XL  130           // halo: x = -1 .. 128
#define AS_PLANE (AS_XL * PADC)
#define AS_ELEMS (3 * AS_PLANE)
#define BS_ROW PADK
#define BS_ELEMS (9 * C_OUT * BS_ROW)

// ---------------------------------------------------------------------------
// Kernel 0a: repack conv weights [64,384,3,3] f32 -> [9][n=64][c=384] bf16
// ---------------------------------------------------------------------------
__global__ void repack_weights_kernel(const float* __restrict__ w,
                                      __bf16* __restrict__ wt) {
  int idx = blockIdx.x * blockDim.x + threadIdx.x;   // over 9*64*384
  if (idx >= 9 * C_OUT * C_IN) return;
  int c    = idx % C_IN;
  int rest = idx / C_IN;
  int n    = rest % C_OUT;
  int khkw = rest / C_OUT;
  int kh = khkw / 3, kw = khkw % 3;
  wt[idx] = (__bf16)w[(((size_t)n * C_IN + c) * 3 + kh) * 3 + kw];
}

// ---------------------------------------------------------------------------
// Kernel 0b: convert input activations f32 -> bf16 (one HBM pass)
// ---------------------------------------------------------------------------
__global__ __launch_bounds__(256) void cvt_input_kernel(
    const float* __restrict__ in, __bf16* __restrict__ out, int total) {
  int i = (blockIdx.x * 256 + threadIdx.x) * 8;
  if (i >= total) return;                  // total is a multiple of 8
  float4 a = *(const float4*)(in + i);
  float4 b = *(const float4*)(in + i + 4);
  v8bf o;
  o[0] = (__bf16)a.x; o[1] = (__bf16)a.y; o[2] = (__bf16)a.z; o[3] = (__bf16)a.w;
  o[4] = (__bf16)b.x; o[5] = (__bf16)b.y; o[6] = (__bf16)b.z; o[7] = (__bf16)b.w;
  *(v8bf*)(out + i) = o;
}

// ---------------------------------------------------------------------------
// Kernel 1: implicit-GEMM 3x3 conv + BN + ReLU, v_wmma_f32_16x16x32_bf16.
// Block = 1 BEV row (128 px) x 64 out-ch. 8 waves; wave = (nt, mh) owns
// 4 m-tiles. Per 32-channel round: stage 3x130 halo + all 9 weight slices,
// then 9 khkw x 4 mt = 36 WMMAs per wave. Next round's global lines are
// prefetched (global_prefetch_b8) before the WMMA burst.
// ---------------------------------------------------------------------------
__global__ __launch_bounds__(256) void conv_wmma_kernel(
    const __bf16* __restrict__ inbf,    // [B,384,128,128] bf16
    const __bf16* __restrict__ wt,      // [9,64,384] bf16
    const float* __restrict__ bn_s,
    const float* __restrict__ bn_b,
    float* __restrict__ bev)            // [B,128,128,64]
{
  __shared__ __bf16 As[AS_ELEMS];       // [kh][xl 0..129][c], c-contiguous
  __shared__ __bf16 Bs[BS_ELEMS];       // [khkw][n][k],      k-contiguous

  const int t   = threadIdx.x;
  const int row = blockIdx.x;           // over B*128
  const int b   = row >> 7;
  const int y   = row & 127;

  const int wave = t >> 5;
  const int lane = t & 31;
  const int nt   = wave >> 1;           // 0..3  channel quarter
  const int mh   = wave & 1;            // 0..1  pixel half (64 px)
  const int hi   = lane >> 4;
  const int lm   = lane & 15;

  // zero the halo columns (xl = 0 and 129) once; never overwritten later
  for (int i = t; i < 3 * KSTEP * 2; i += 256) {   // 192 slots
    int kh = i / (KSTEP * 2);
    int r  = i % (KSTEP * 2);
    int c  = r >> 1;
    int xl = (r & 1) ? (AS_XL - 1) : 0;
    As[kh * AS_PLANE + xl * PADC + c] = (__bf16)0.f;
  }

  v8f acc[4] = {{}, {}, {}, {}};

  // per-wave LDS read bases (in halves)
  const int bs_n   = (nt * 16 + lm) * BS_ROW;          // + khkw*64*BS_ROW
  const int as_ch  = hi * 8;                           // first ch chunk

#pragma unroll 1
  for (int kc0 = 0; kc0 < C_IN; kc0 += KSTEP) {
    // ---- stage A: 3 rows x 128 px x 32 ch (transpose x<->c into LDS)
#pragma unroll
    for (int j = 0; j < 24; ++j) {
      int idx = t + j * 256;            // 0..6143 = (3*32 rows) * 64 uints
      int u   = idx & 63;               // x pair: ix = 2u, 2u+1
      int r   = idx >> 6;               // 0..95
      int kh  = r >> 5;
      int c   = r & 31;
      int iy  = y + kh - 1;
      unsigned v = 0;
      if (iy >= 0 && iy < HW)
        v = *(const unsigned*)(inbf +
              (((size_t)b * C_IN + (kc0 + c)) * HW + iy) * HW + 2 * u);
      union { unsigned u32; __bf16 h[2]; } cv; cv.u32 = v;
      __bf16* dst = &As[kh * AS_PLANE + (2 * u + 1) * PADC + c];
      dst[0]    = cv.h[0];              // xl = 2u+1
      dst[PADC] = cv.h[1];              // xl = 2u+2
    }
    // ---- stage B: 9 x 64 x 32 bf16, b128 both sides
#pragma unroll
    for (int j = 0; j < 9; ++j) {
      int idx = t + j * 256;            // 0..2303
      int k8  = idx & 3;                // 8-ch chunk
      int r   = idx >> 2;               // (khkw*64 + n) : 0..575
      v8bf wv = *(const v8bf*)(wt + (size_t)r * C_IN + kc0 + k8 * 8);
      *(v8bf*)&Bs[r * BS_ROW + k8 * 8] = wv;
    }
    __syncthreads();

    // ---- prefetch next round's global lines while WMMAs run
    const int kc1 = kc0 + KSTEP;
    if (kc1 < C_IN) {
      {
        // next A slice: 96 (kh,c) rows -> one line per row, threads 0..95
        int r  = t;
        if (r < 96) {
          int kh = r >> 5, c = r & 31;
          int iy = y + kh - 1;
          if (iy >= 0 && iy < HW)
            __builtin_prefetch(inbf +
                (((size_t)b * C_IN + (kc1 + c)) * HW + iy) * HW, 0, 3);
        }
      }
      {
        // next B slice: 576 rows of 64B -> one line per 2 rows
        int r = t + 256;                 // threads cover rows 256..511 & below
        __builtin_prefetch(wt + (size_t)(t * 2) * C_IN + kc1, 0, 3);
        __builtin_prefetch(wt + (size_t)(r)     * C_IN + kc1, 0, 3);
      }
    }

    // ---- 9 x 4 WMMAs out of LDS
#pragma unroll
    for (int khkw = 0; khkw < 9; ++khkw) {
      const int kh = khkw / 3, kw = khkw % 3;
      const __bf16* bp = &Bs[khkw * C_OUT * BS_ROW + bs_n + hi * 8];
      v8bf blo = *(const v8bf*)bp;
      v8bf bhi = *(const v8bf*)(bp + 16);
      v16bf bfrag = __builtin_shufflevector(blo, bhi,
          0,1,2,3,4,5,6,7,8,9,10,11,12,13,14,15);
#pragma unroll
      for (int mt = 0; mt < 4; ++mt) {
        const int m  = mh * 64 + mt * 16 + lm;
        const int xl = m + kw;          // 0..129
        const __bf16* ap = &As[kh * AS_PLANE + xl * PADC + as_ch];
        v8bf alo = *(const v8bf*)ap;         // ch hi*8   .. hi*8+7
        v8bf ahi = *(const v8bf*)(ap + 16);  // ch 16+hi*8.. 16+hi*8+7
        v16bf afrag = __builtin_shufflevector(alo, ahi,
            0,1,2,3,4,5,6,7,8,9,10,11,12,13,14,15);
        acc[mt] = __builtin_amdgcn_wmma_f32_16x16x32_bf16(
                      false, afrag, false, bfrag, (short)0, acc[mt],
                      false, false);
      }
    }
    __syncthreads();
  }

  // ---- epilogue: BN + ReLU, channel-contiguous BEV store
  const int n = nt * 16 + lm;
  const float s  = bn_s[n];
  const float bb = bn_b[n];
#pragma unroll
  for (int mt = 0; mt < 4; ++mt) {
#pragma unroll
    for (int r = 0; r < 8; ++r) {
      int m = mh * 64 + mt * 16 + r + hi * 8;
      float v = acc[mt][r] * s + bb;
      v = v > 0.f ? v : 0.f;
      bev[((size_t)row * HW + m) * C_OUT + n] = v;
    }
  }
}

// ---------------------------------------------------------------------------
// Kernel 2: per-point bilinear gather (L2-resident BEV) + 3 heads + softmax
// correction + concat. One thread per point.
// ---------------------------------------------------------------------------
__global__ __launch_bounds__(256) void points_kernel(
    const float* __restrict__ pts,      // [N,9]
    const float* __restrict__ bev,      // [B,128,128,64]
    const float* __restrict__ seg_w, const float* __restrict__ seg_b,
    const float* __restrict__ reg_w, const float* __restrict__ reg_b,
    const float* __restrict__ emb_w, const float* __restrict__ emb_b,
    float* __restrict__ out, int N)
{
  __shared__ float Wh[8][64];
  __shared__ float Bh[8];
  for (int i = threadIdx.x; i < 512; i += 256) {
    int r = i >> 6, c = i & 63;
    float v = (r < 3) ? seg_w[r * 64 + c]
            : (r < 6) ? reg_w[(r - 3) * 64 + c]
                      : emb_w[(r - 6) * 64 + c];
    Wh[r][c] = v;
  }
  if (threadIdx.x < 8) {
    int r = threadIdx.x;
    Bh[r] = (r < 3) ? seg_b[r] : (r < 6) ? reg_b[r - 3] : emb_b[r - 6];
  }
  __syncthreads();

  int n = blockIdx.x * 256 + threadIdx.x;
  if (n >= N) return;

  float p[9];
#pragma unroll
  for (int j = 0; j < 9; ++j) p[j] = pts[(size_t)n * 9 + j];

  int   b  = (int)p[0];
  float px = (p[1] + 51.2f) * 1.25f;    // /(0.1*8)
  float py = (p[2] + 51.2f) * 1.25f;
  float fx = floorf(px), fy = floorf(py);
  int x0 = min(max((int)fx,     0), HW - 1);
  int x1 = min(max((int)fx + 1, 0), HW - 1);
  int y0 = min(max((int)fy,     0), HW - 1);
  int y1 = min(max((int)fy + 1, 0), HW - 1);
  float x0f = (float)x0, x1f = (float)x1, y0f = (float)y0, y1f = (float)y1;
  float wa = (x1f - px) * (y1f - py);
  float wb = (x1f - px) * (py - y0f);
  float wc = (px - x0f) * (y1f - py);
  float wd = (px - x0f) * (py - y0f);

  const float* base = bev + (size_t)b * HW * HW * C_OUT;
  const float4* pa = (const float4*)(base + ((size_t)y0 * HW + x0) * C_OUT);
  const float4* pb = (const float4*)(base + ((size_t)y1 * HW + x0) * C_OUT);
  const float4* pc = (const float4*)(base + ((size_t)y0 * HW + x1) * C_OUT);
  const float4* pd = (const float4*)(base + ((size_t)y1 * HW + x1) * C_OUT);

  float acc[8];
#pragma unroll
  for (int r = 0; r < 8; ++r) acc[r] = Bh[r];

  for (int c4 = 0; c4 < 16; ++c4) {
    float4 fa = pa[c4], fb = pb[c4], fc = pc[c4], fd = pd[c4];
    float f0 = fa.x * wa + fb.x * wb + fc.x * wc + fd.x * wd;
    float f1 = fa.y * wa + fb.y * wb + fc.y * wc + fd.y * wd;
    float f2 = fa.z * wa + fb.z * wb + fc.z * wc + fd.z * wd;
    float f3 = fa.w * wa + fb.w * wb + fc.w * wc + fd.w * wd;
    int c = c4 * 4;
#pragma unroll
    for (int r = 0; r < 8; ++r)
      acc[r] += f0 * Wh[r][c] + f1 * Wh[r][c + 1]
              + f2 * Wh[r][c + 2] + f3 * Wh[r][c + 3];
  }

  float l0 = acc[0], l1 = acc[1], l2 = acc[2];
  float mx = fmaxf(l0, fmaxf(l1, l2));
  float e0 = __expf(l0 - mx), e1 = __expf(l1 - mx), e2 = __expf(l2 - mx);
  float s  = e0 + e1 + e2;
  int   ci = (l0 >= l1 && l0 >= l2) ? 0 : (l1 >= l2 ? 1 : 2);
  float pmax = fmaxf(e0, fmaxf(e1, e2)) / s;
  bool  mask = (pmax > 0.5f) && (ci == 2);
  float ox = mask ? acc[3] : 0.f;
  float oy = mask ? acc[4] : 0.f;
  float oz = mask ? acc[5] : 0.f;

  float* o = out + (size_t)n * 17;
  o[0] = p[0];
  o[1] = p[1] + ox; o[2] = p[2] + oy; o[3] = p[3] + oz;
  o[4] = p[4]; o[5] = p[5]; o[6] = p[6]; o[7] = p[7]; o[8] = p[8];
  o[9]  = l0; o[10] = l1; o[11] = l2;
  o[12] = acc[3]; o[13] = acc[4]; o[14] = acc[5];
  o[15] = acc[6]; o[16] = acc[7];
}

// ---------------------------------------------------------------------------
extern "C" void kernel_launch(void* const* d_in, const int* in_sizes, int n_in,
                              void* d_out, int out_size, void* d_ws, size_t ws_size,
                              hipStream_t stream) {
  const float* points = (const float*)d_in[0];
  const float* sf2d   = (const float*)d_in[1];
  const float* conv_w = (const float*)d_in[2];
  const float* bn_s   = (const float*)d_in[3];
  const float* bn_b   = (const float*)d_in[4];
  const float* seg_w  = (const float*)d_in[5];
  const float* seg_b  = (const float*)d_in[6];
  const float* reg_w  = (const float*)d_in[7];
  const float* reg_b  = (const float*)d_in[8];
  const float* emb_w  = (const float*)d_in[9];
  const float* emb_b  = (const float*)d_in[10];
  float* out = (float*)d_out;

  const int N  = in_sizes[0] / 9;
  const int B  = in_sizes[1] / (C_IN * HW * HW);
  const int in_total = in_sizes[1];

  // workspace layout:
  //   [0, 442KB)        repacked bf16 weights
  //   [1MB, 1MB+50.4MB) bf16 input activations
  //   [64MB, 64MB+16MB) BEV f32
  __bf16* wt   = (__bf16*)d_ws;
  __bf16* inbf = (__bf16*)((char*)d_ws + (1u << 20));
  float*  bev  = (float*)((char*)d_ws + (64u << 20));

  {
    int total = 9 * C_OUT * C_IN;
    repack_weights_kernel<<<(total + 255) / 256, 256, 0, stream>>>(conv_w, wt);
  }
  cvt_input_kernel<<<(in_total / 8 + 255) / 256, 256, 0, stream>>>(
      sf2d, inbf, in_total);
  conv_wmma_kernel<<<B * HW, 256, 0, stream>>>(inbf, wt, bn_s, bn_b, bev);
  points_kernel<<<(N + 255) / 256, 256, 0, stream>>>(
      points, bev, seg_w, seg_b, reg_w, reg_b, emb_w, emb_b, out, N);
}
